// SelfAttention_31542239822466
// MI455X (gfx1250) — matrix operand order repaired and checked by reading, not compiled, not run on hardware
//
#include <hip/hip_runtime.h>
#include <hip/hip_bf16.h>

// Problem constants (match reference setup_inputs)
#define BB 4
#define SS 4096
#define DD 512
#define HH 256

typedef __attribute__((ext_vector_type(16))) __bf16 v16bf;
typedef __attribute__((ext_vector_type(8)))  __bf16 v8bf;
typedef __attribute__((ext_vector_type(8)))  float  v8f;

#define WMMA_BF16(acc, a, b) \
  __builtin_amdgcn_wmma_f32_16x16x32_bf16(false, (a), false, (b), (short)0, (acc), false, false)

// ---------------------------------------------------------------------------
// Kernel 1: one-shot f32->bf16 conversion of x, plus transposed bf16 copies of
// the weights so every WMMA A/B operand downstream is a contiguous 16B load.
//   xb  : [B*S][D]   (A operand for projections)
//   wqT : [H][D], wkT : [H][D], wvT : [D][D]   (B operands, column-contiguous)
// ---------------------------------------------------------------------------
__global__ __launch_bounds__(256) void prep_convert(
    const float* __restrict__ x,
    const float* __restrict__ Wq, const float* __restrict__ Wk,
    const float* __restrict__ Wv,
    __bf16* __restrict__ xb,
    __bf16* __restrict__ wqT, __bf16* __restrict__ wkT, __bf16* __restrict__ wvT) {
  const int XN = BB * SS * DD;      // 8388608
  const int QK = DD * HH;           // 131072
  int i = blockIdx.x * 256 + threadIdx.x;
  if (i < XN) { xb[i] = (__bf16)x[i]; return; }
  int j = i - XN;
  if (j < QK) {
    int d = j / HH, h = j % HH;
    wqT[h * DD + d] = (__bf16)Wq[j];
  } else if (j < 2 * QK) {
    int jj = j - QK;
    int d = jj / HH, h = jj % HH;
    wkT[h * DD + d] = (__bf16)Wk[jj];
  } else if (j < 2 * QK + DD * DD) {
    int jj = j - 2 * QK;
    int d = jj / DD, d2 = jj % DD;
    wvT[d2 * DD + d] = (__bf16)Wv[jj];
  }
}

// ---------------------------------------------------------------------------
// Kernel 2: fused QKV projection, pure-bf16 operands. One wave = one 16x16
// tile of the concatenated [B*S, H|H|D] output, K = D = 512 (16 WMMA steps).
// q,k written row-major bf16; v written transposed (vt[b][d][s]) so the
// attn@v B-operand is contiguous.
// ---------------------------------------------------------------------------
__global__ __launch_bounds__(256) void proj_kernel(
    const __bf16* __restrict__ xb,
    const float* __restrict__ bq, const float* __restrict__ bk,
    const float* __restrict__ bv,
    const __bf16* __restrict__ wqT, const __bf16* __restrict__ wkT,
    const __bf16* __restrict__ wvT,
    __bf16* __restrict__ q, __bf16* __restrict__ k, __bf16* __restrict__ vt) {
  const int lane  = threadIdx.x & 31;
  const int wave  = threadIdx.x >> 5;
  const int m     = lane & 15;        // A row / C column index
  const int lhalf = lane >> 4;        // K-half this lane provides
  const int aOff  = lhalf ? 16 : 0;   // A layout: lanes>=16 start at K=16

  const int colTile = blockIdx.y * 8 + wave;   // 0..63 over 1024 fused cols
  const int col0    = colTile * 16;
  const int row0    = blockIdx.x * 16;         // over B*S rows

  const __bf16* WT; const float* bias; int n0;
  if (col0 < HH)            { WT = wqT; bias = bq; n0 = col0; }
  else if (col0 < 2 * HH)   { WT = wkT; bias = bk; n0 = col0 - HH; }
  else                      { WT = wvT; bias = bv; n0 = col0 - 2 * HH; }

  const int n = n0 + m;                                  // output column (lane)
  const __bf16* arow = xb + (size_t)(row0 + m) * DD + aOff;
  const __bf16* brow = WT + (size_t)n * DD + lhalf * 16;

  v8f acc = {};
#pragma unroll 4
  for (int k0 = 0; k0 < DD; k0 += 32) {
    v16bf a;
    *(v8bf*)&a       = *(const v8bf*)(arow + k0);        // K = aOff..aOff+7
    *((v8bf*)&a + 1) = *(const v8bf*)(arow + k0 + 8);    // K = aOff+8..+15
    v16bf bm = *(const v16bf*)(brow + k0);               // 32B aligned
    acc = WMMA_BF16(acc, a, bm);
  }

  const float bval = bias[n];
  if (col0 < 2 * HH) {
    __bf16* dst = (col0 < HH) ? q : k;            // row-major [B*S][H]
#pragma unroll
    for (int r = 0; r < 8; ++r) {
      int M = r + lhalf * 8;
      dst[(size_t)(row0 + M) * HH + n] = (__bf16)(acc[r] + bval);
    }
  } else {                                        // transposed vt[b][d][s]
    int gb = row0 / SS, srow = row0 % SS;         // 16-row tile never crosses b
#pragma unroll
    for (int r = 0; r < 8; ++r) {
      int M = r + lhalf * 8;
      vt[((size_t)gb * DD + n) * SS + (srow + M)] = (__bf16)(acc[r] + bval);
    }
  }
}

// ---------------------------------------------------------------------------
// Kernel 3: fused attention for one (batch, 16-row q tile):
//   stage 1: scores[16][4096] = (q @ k^T)/16 via WMMA -> LDS (256 KB);
//            the 16xH q A-fragment is hoisted (loaded once, 32 tiles reuse it)
//   stage 2: exact row softmax + fuzzy clamp; write attn f32 to d_out and
//            pack bf16 fmf values into the low half of each thread's own
//            1KB score chunk (race-free in-place re-use of the LDS strip)
//   stage 3: out[16][512] = attn @ v via WMMA; 4 accumulators per wave so
//            each LDS A-load feeds 4 independent WMMA chains
// ---------------------------------------------------------------------------
__global__ __launch_bounds__(256) void attn_kernel(
    const __bf16* __restrict__ q, const __bf16* __restrict__ k,
    const __bf16* __restrict__ vt,
    float* __restrict__ out, float* __restrict__ attn) {
  extern __shared__ float sc[];          // 16*SS scores + 16*16 reduction
  float* red = sc + 16 * SS;

  const int lane  = threadIdx.x & 31;
  const int wave  = threadIdx.x >> 5;
  const int m     = lane & 15;
  const int lhalf = lane >> 4;
  const int aOff  = lhalf ? 16 : 0;

  const int b     = blockIdx.x / (SS / 16);
  const int qrow0 = (blockIdx.x % (SS / 16)) * 16;

  // ---- Stage 1: scores -> LDS ----
  {
    const __bf16* qrow = q + ((size_t)b * SS + qrow0 + m) * HH + aOff;
    v16bf afrag[8];                                  // 16xH A tile, loaded once
#pragma unroll
    for (int st = 0; st < 8; ++st) {
      v16bf a;
      *(v8bf*)&a       = *(const v8bf*)(qrow + st * 32);
      *((v8bf*)&a + 1) = *(const v8bf*)(qrow + st * 32 + 8);
      afrag[st] = a;
    }
    for (int t = 0; t < 32; ++t) {
      const int c0 = wave * 512 + t * 16;            // k-column tile base
      const __bf16* krow = k + ((size_t)b * SS + c0 + m) * HH + lhalf * 16;
      v8f acc = {};
#pragma unroll
      for (int st = 0; st < 8; ++st) {               // 8 back-to-back WMMAs
        v16bf bm = *(const v16bf*)(krow + st * 32);
        acc = WMMA_BF16(acc, afrag[st], bm);
      }
#pragma unroll
      for (int r = 0; r < 8; ++r)
        sc[(size_t)(r + lhalf * 8) * SS + (c0 + m)] = acc[r] * 0.0625f; // 1/sqrt(H)
    }
  }
  __syncthreads();

  // ---- Stage 2: softmax + fuzzy membership; 16 threads per row ----
  {
    const int r  = threadIdx.x >> 4;       // row 0..15
    const int t  = threadIdx.x & 15;       // 256-col chunk per thread
    float* srow  = sc + (size_t)r * SS;
    const int cs = t * 256;
    const int st4 = t * 4;                 // bank stagger for read-only scans

    float mx = -3.0e38f;
    for (int c = 0; c < 256; ++c)
      mx = fmaxf(mx, srow[cs + ((c + st4) & 255)]);
    red[r * 16 + t] = mx;
    __syncthreads();
    float rowmax = -3.0e38f;
    for (int i = 0; i < 16; ++i) rowmax = fmaxf(rowmax, red[r * 16 + i]);

    float sum = 0.f;
    for (int c = 0; c < 256; ++c)
      sum += __expf(srow[cs + ((c + st4) & 255)] - rowmax);
    __syncthreads();                        // all done reading maxes
    red[r * 16 + t] = sum;
    __syncthreads();
    float tot = 0.f;
    for (int i = 0; i < 16; ++i) tot += red[r * 16 + i];
    const float inv   = 1.f / tot;
    const float inv03 = 1.f / 0.3f;         // 1/(b-a) == 1/(d-c)

    // Pack bf16 fmf values into the low 512B of this thread's own 1KB chunk:
    // writing element c/2 (<= c, already consumed) -> race-free in place.
    __bf16* packw = (__bf16*)sc + (size_t)r * (SS * 2) + t * 512;
    float*  arow  = attn + ((size_t)b * SS + qrow0 + r) * SS;
    for (int c = 0; c < 256; ++c) {         // sequential: required for safety
      float a = __expf(srow[cs + c] - rowmax) * inv;
      float f = fmaxf(fminf(a * inv03, (1.f - a) * inv03), 0.f);
      arow[cs + c]  = f;                    // attn output (f32)
      packw[c]      = (__bf16)f;            // A operand for stage 3
    }
  }
  __syncthreads();

  // ---- Stage 3: out = attn @ v; wave owns 64 output cols = 4 C tiles.
  // One LDS A-load per K-step feeds 4 independent WMMA accumulator chains. ----
  {
    const __bf16* scb = (const __bf16*)sc;  // packed-chunk bf16 attn
    const int n0 = wave * 64;
    const __bf16* vrow = vt + ((size_t)b * DD + n0 + m) * SS + lhalf * 16;
    const size_t cstr = (size_t)16 * SS;    // 16 v-columns ahead
    v8f acc0 = {}, acc1 = {}, acc2 = {}, acc3 = {};
    for (int s0 = 0; s0 < SS; s0 += 32) {
      __builtin_prefetch((const void*)(vrow + s0 + 1024), 0, 0); // stream v ahead
      const int base = m * (SS * 2) + ((s0 >> 8) << 9) + (s0 & 255) + aOff;
      v16bf a;
      *(v8bf*)&a       = *(const v8bf*)(scb + base);
      *((v8bf*)&a + 1) = *(const v8bf*)(scb + base + 8);
      v16bf b0 = *(const v16bf*)(vrow + s0);
      v16bf b1 = *(const v16bf*)(vrow + cstr + s0);
      v16bf b2 = *(const v16bf*)(vrow + 2 * cstr + s0);
      v16bf b3 = *(const v16bf*)(vrow + 3 * cstr + s0);
      acc0 = WMMA_BF16(acc0, a, b0);
      acc1 = WMMA_BF16(acc1, a, b1);
      acc2 = WMMA_BF16(acc2, a, b2);
      acc3 = WMMA_BF16(acc3, a, b3);
    }
    float* orow = out + ((size_t)b * SS + qrow0) * DD + n0 + m;
#pragma unroll
    for (int r = 0; r < 8; ++r) {
      float* p = orow + (size_t)(r + lhalf * 8) * DD;
      p[0]  = acc0[r];
      p[16] = acc1[r];
      p[32] = acc2[r];
      p[48] = acc3[r];
    }
  }
}

// ---------------------------------------------------------------------------
extern "C" void kernel_launch(void* const* d_in, const int* in_sizes, int n_in,
                              void* d_out, int out_size, void* d_ws, size_t ws_size,
                              hipStream_t stream) {
  const float* x  = (const float*)d_in[0];
  const float* Wq = (const float*)d_in[1];
  const float* bq = (const float*)d_in[2];
  const float* Wk = (const float*)d_in[3];
  const float* bk = (const float*)d_in[4];
  const float* Wv = (const float*)d_in[5];
  const float* bv = (const float*)d_in[6];

  float* out  = (float*)d_out;                       // [B,S,D]
  float* attn = out + (size_t)BB * SS * DD;          // [B,S,S]

  // Workspace layout (bf16), all offsets 32B-aligned:
  __bf16* wqT = (__bf16*)d_ws;                       // [H][D]
  __bf16* wkT = wqT + (size_t)DD * HH;               // [H][D]
  __bf16* wvT = wkT + (size_t)DD * HH;               // [D][D]
  __bf16* xb  = wvT + (size_t)DD * DD;               // [B*S][D]
  __bf16* qb  = xb  + (size_t)BB * SS * DD;          // [B*S][H]
  __bf16* kb  = qb  + (size_t)BB * SS * HH;          // [B*S][H]
  __bf16* vtb = kb  + (size_t)BB * SS * HH;          // [B][D][S]

  const int pN = BB * SS * DD + 2 * DD * HH + DD * DD;
  prep_convert<<<(pN + 255) / 256, 256, 0, stream>>>(x, Wq, Wk, Wv,
                                                     xb, wqT, wkT, wvT);

  dim3 pgrid((BB * SS) / 16, 8);                     // 1024 row tiles x 64 col tiles
  proj_kernel<<<pgrid, 256, 0, stream>>>(xb, bq, bk, bv, wqT, wkT, wvT, qb, kb, vtb);

  const size_t lds_bytes = (16 * SS + 16 * 16) * sizeof(float); // 257 KB
  attn_kernel<<<(BB * SS) / 16, 256, lds_bytes, stream>>>(qb, kb, vtb, out, attn);
}
